// BSMSProcessor_85658827751892
// MI455X (gfx1250) — matrix operand order repaired and checked by reading, not compiled, not run on hardware
//
#include <hip/hip_runtime.h>
#include <hip/hip_bf16.h>

typedef __attribute__((ext_vector_type(16))) _Float16 v16h;
typedef __attribute__((ext_vector_type(8)))  float    v8f;

// ---------------------------------------------------------------------------
// Wave-level 16xKx64 GEMM using v_wmma_f32_16x16x32_f16.
// Input tile in LDS (f16, row-major, leading dim ldk), weights pre-packed as
// per-lane B fragments (chunk-major: [chunk][ntile][lane][16]), f32 accumulate.
// ---------------------------------------------------------------------------
__device__ __forceinline__ void gemm16x64(const _Float16* in, int ldk, int nCh,
                                          const _Float16* __restrict__ wf,
                                          const float* __restrict__ bias,
                                          bool relu, _Float16* outH, float* outF,
                                          int lane) {
  v8f a0 = {}, a1 = {}, a2 = {}, a3 = {};
  const int M = lane & 15;
  const int half = lane >> 4;
  const _Float16* rp = in + M * ldk;
  for (int ch = 0; ch < nCh; ++ch) {
    // Assemble A fragment per CDNA5 16-bit A layout:
    // lanes 0-15: rows M, K pairs {2v, 2v+1} (v<4) / {16+2(v-4)} ; lanes 16-31: +8
    v16h av;
#pragma unroll
    for (int p = 0; p < 8; ++p) {
      int kp = ch * 32 + ((p < 4) ? (half * 8 + 2 * p) : (16 + half * 8 + 2 * (p - 4)));
      av[2 * p]     = rp[kp];
      av[2 * p + 1] = rp[kp + 1];
    }
    const _Float16* wb = wf + ch * 2048 + lane * 16;
    if (ch + 1 < nCh) __builtin_prefetch(wb + 2048, 0, 1);
    v16h b0 = *(const v16h*)(wb);
    v16h b1 = *(const v16h*)(wb + 512);
    v16h b2 = *(const v16h*)(wb + 1024);
    v16h b3 = *(const v16h*)(wb + 1536);
    a0 = __builtin_amdgcn_wmma_f32_16x16x32_f16(false, av, false, b0, (short)0, a0, false, false);
    a1 = __builtin_amdgcn_wmma_f32_16x16x32_f16(false, av, false, b1, (short)0, a1, false, false);
    a2 = __builtin_amdgcn_wmma_f32_16x16x32_f16(false, av, false, b2, (short)0, a2, false, false);
    a3 = __builtin_amdgcn_wmma_f32_16x16x32_f16(false, av, false, b3, (short)0, a3, false, false);
  }
  // C layout: VGPR r -> row (half*8 + r), col = ntile*16 + (lane&15)
#pragma unroll
  for (int r = 0; r < 8; ++r) {
    int m = half * 8 + r;
    float v0 = a0[r] + bias[M];
    float v1 = a1[r] + bias[16 + M];
    float v2 = a2[r] + bias[32 + M];
    float v3 = a3[r] + bias[48 + M];
    if (relu) {
      v0 = fmaxf(v0, 0.f); v1 = fmaxf(v1, 0.f);
      v2 = fmaxf(v2, 0.f); v3 = fmaxf(v3, 0.f);
      outH[m * 64 + M]      = (_Float16)v0;
      outH[m * 64 + 16 + M] = (_Float16)v1;
      outH[m * 64 + 32 + M] = (_Float16)v2;
      outH[m * 64 + 48 + M] = (_Float16)v3;
    } else {
      outF[m * 64 + M]      = v0;
      outF[m * 64 + 16 + M] = v1;
      outF[m * 64 + 32 + M] = v2;
      outF[m * 64 + 48 + M] = v3;
    }
  }
}

// 3-layer MLP on one 16-row tile: A(f16,16xldk) -> Bh(f16,16x64) -> A -> C(f32,16x64)
__device__ __forceinline__ void mlp3(_Float16* A, int ldk, int c0, _Float16* Bh, float* C,
                                     const _Float16* w0, const float* b0,
                                     const _Float16* w1, const float* b1,
                                     const _Float16* w2, const float* b2,
                                     const float* gain, const float* beta, int lane) {
  gemm16x64(A, ldk, c0, w0, b0, true,  Bh, nullptr, lane);
  gemm16x64(Bh, 64, 2,  w1, b1, true,  A,  nullptr, lane);
  gemm16x64(A, 64, 2,   w2, b2, false, nullptr, C, lane);
  if (gain != nullptr && lane < 16) {   // LayerNorm over 64 features, row per lane
    float* row = C + lane * 64;
    float mu = 0.f;
#pragma unroll 8
    for (int c = 0; c < 64; ++c) mu += row[c];
    mu *= (1.f / 64.f);
    float var = 0.f;
#pragma unroll 8
    for (int c = 0; c < 64; ++c) { float d = row[c] - mu; var += d * d; }
    var *= (1.f / 64.f);
    float rs = rsqrtf(var + 1e-5f);
#pragma unroll 8
    for (int c = 0; c < 64; ++c) row[c] = (row[c] - mu) * rs * gain[c] + beta[c];
  }
}

// ---------------------------------------------------------------------------
// Edge message kernel: gather [x_i | x_j | dpos | ||dpos||], edge MLP (WMMA),
// LN, atomic scatter-add into aggr.
// ---------------------------------------------------------------------------
__global__ __launch_bounds__(128)
void k_edge_gmp(const float* __restrict__ h, const float* __restrict__ ps,
                const int* __restrict__ gi, const int* __restrict__ gj, int E,
                const _Float16* __restrict__ w0, const float* __restrict__ b0,
                const _Float16* __restrict__ w1, const float* __restrict__ b1,
                const _Float16* __restrict__ w2, const float* __restrict__ b2,
                const float* __restrict__ gain, const float* __restrict__ beta,
                float* __restrict__ aggr) {
  __shared__ _Float16 shA[4][16 * 160];
  __shared__ _Float16 shB[4][16 * 64];
  __shared__ float    shC[4][16 * 64];
  const int wv = threadIdx.x >> 5, lane = threadIdx.x & 31;
  const int e0 = (blockIdx.x * 4 + wv) * 16;
  if (e0 >= E) return;
  _Float16* A = shA[wv];
  _Float16* Bh = shB[wv];
  float* C = shC[wv];

  int iIdx = 0, jIdx = 0;
  { int me = e0 + (lane & 15); if (me < E) { iIdx = gi[me]; jIdx = gj[me]; } }

  for (int t = lane; t < 16 * 64; t += 32) {
    int r = t >> 6, c = t & 63;
    int ii = __shfl(iIdx, r, 32);
    int jj = __shfl(jIdx, r, 32);
    bool ok = (e0 + r) < E;
    float xi = ok ? h[(size_t)ii * 64 + c] : 0.f;
    float xj = ok ? h[(size_t)jj * 64 + c] : 0.f;
    A[r * 160 + c]      = (_Float16)xi;
    A[r * 160 + 64 + c] = (_Float16)xj;
  }
  for (int t = lane; t < 16 * 32; t += 32) {
    int r = t >> 5, c = t & 31;
    A[r * 160 + 128 + c] = (_Float16)0.f;
  }
  if (lane < 16 && (e0 + lane) < E) {
    float dx = ps[(size_t)iIdx * 3 + 0] - ps[(size_t)jIdx * 3 + 0];
    float dy = ps[(size_t)iIdx * 3 + 1] - ps[(size_t)jIdx * 3 + 1];
    float dz = ps[(size_t)iIdx * 3 + 2] - ps[(size_t)jIdx * 3 + 2];
    A[lane * 160 + 128] = (_Float16)dx;
    A[lane * 160 + 129] = (_Float16)dy;
    A[lane * 160 + 130] = (_Float16)dz;
    A[lane * 160 + 131] = (_Float16)sqrtf(dx * dx + dy * dy + dz * dz);
  }

  mlp3(A, 160, 5, Bh, C, w0, b0, w1, b1, w2, b2, gain, beta, lane);

  for (int t = lane; t < 16 * 64; t += 32) {
    int r = t >> 6, c = t & 63;
    int ii = __shfl(iIdx, r, 32);
    if ((e0 + r) < E) atomicAdd(&aggr[(size_t)ii * 64 + c], C[r * 64 + c]);
  }
}

// ---------------------------------------------------------------------------
// Generic dense 3-layer MLP over rows: in = [inA | inB | pad], optional
// residual, optional LN, arbitrary narrow output (decoder). WMMA engine.
// ---------------------------------------------------------------------------
__global__ __launch_bounds__(128)
void k_dense_mlp(const float* __restrict__ inA, int colsA,
                 const float* __restrict__ inB, int colsB,
                 const float* __restrict__ resid, float* __restrict__ out,
                 int outCols, int N,
                 const _Float16* __restrict__ w0, const float* __restrict__ b0,
                 int c0, int ldk,
                 const _Float16* __restrict__ w1, const float* __restrict__ b1,
                 const _Float16* __restrict__ w2, const float* __restrict__ b2,
                 const float* __restrict__ gain, const float* __restrict__ beta) {
  __shared__ _Float16 shA[4][16 * 160];
  __shared__ _Float16 shB[4][16 * 64];
  __shared__ float    shC[4][16 * 64];
  const int wv = threadIdx.x >> 5, lane = threadIdx.x & 31;
  const int r0 = (blockIdx.x * 4 + wv) * 16;
  if (r0 >= N) return;
  _Float16* A = shA[wv];
  _Float16* Bh = shB[wv];
  float* C = shC[wv];

  const int tot = 16 * ldk;
  for (int t = lane; t < tot; t += 32) {
    int r = t / ldk, c = t - r * ldk;
    int row = r0 + r;
    float v = 0.f;
    if (row < N) {
      if (c < colsA) v = inA[(size_t)row * colsA + c];
      else if (c < colsA + colsB) v = inB[(size_t)row * colsB + (c - colsA)];
    }
    A[r * ldk + c] = (_Float16)v;
  }

  mlp3(A, ldk, c0, Bh, C, w0, b0, w1, b1, w2, b2, gain, beta, lane);

  for (int t = lane; t < 16 * outCols; t += 32) {
    int r = t / outCols, c = t - r * outCols;
    int row = r0 + r;
    if (row < N) {
      float v = C[r * 64 + c];
      if (resid) v += resid[(size_t)row * 64 + c];
      out[(size_t)row * outCols + c] = v;
    }
  }
}

// ---------------------------------------------------------------------------
// Weight packing: fp32 [Kreal x Nreal] -> f16 fragments [chunk][ntile][lane][16]
// matching the per-lane v16h B loads in gemm16x64 (K padded to 32, N to 64).
// ---------------------------------------------------------------------------
__global__ void k_pack_w(const float* __restrict__ W, _Float16* __restrict__ dst,
                         int Kreal, int Nreal, int nCh) {
  int t = blockIdx.x * 256 + threadIdx.x;
  int total = nCh * 2048;
  if (t >= total) return;
  int e  = t & 15;
  int ln = (t >> 4) & 31;
  int nt = (t >> 9) & 3;
  int ch = t >> 11;
  int k = ch * 32 + ((e < 8) ? (8 * (ln >> 4) + e) : (16 + 8 * (ln >> 4) + (e - 8)));
  int n = nt * 16 + (ln & 15);
  float v = (k < Kreal && n < Nreal) ? W[(size_t)k * Nreal + n] : 0.f;
  dst[t] = (_Float16)v;
}

__global__ void k_pack_b(const float* __restrict__ b, float* __restrict__ dst, int Nreal) {
  int t = threadIdx.x;
  dst[t] = (t < Nreal) ? b[t] : 0.f;
}

// ------------------------- graph plumbing kernels --------------------------
__global__ void k_fillv(float* p, int n, float v) {
  int t = blockIdx.x * 256 + threadIdx.x;
  if (t < n) p[t] = v;
}
__global__ void k_deg(const int* gi, int E, float* deg) {
  int t = blockIdx.x * 256 + threadIdx.x;
  if (t < E) atomicAdd(&deg[gi[t]], 1.f);
}
__global__ void k_normed(const float* w, const float* deg, float* nrm, int n) {
  int t = blockIdx.x * 256 + threadIdx.x;
  if (t < n) nrm[t] = w[t] / fmaxf(deg[t], 1.f);
}
__global__ void k_aggrw(const int* gi, const int* gj, const float* nrm, float* aw, int E) {
  int t = blockIdx.x * 256 + threadIdx.x;
  if (t < E) atomicAdd(&aw[gi[t]], nrm[gj[t]]);
}
__global__ void k_ec(const int* gi, const int* gj, const float* nrm, const float* aw,
                     float* ec, int E) {
  int t = blockIdx.x * 256 + threadIdx.x;
  if (t < E) ec[t] = nrm[gj[t]] / (aw[gi[t]] + 1e-12f);
}
__global__ void k_wnew(const float* aw, float* w, int n) {
  int t = blockIdx.x * 256 + threadIdx.x;
  if (t < n) w[t] = aw[t] + 1e-12f;
}
__global__ void k_wconv(const float* __restrict__ x, const int* __restrict__ src,
                        const int* __restrict__ dst, const float* __restrict__ ec,
                        float* __restrict__ out, int E, int C) {
  int t = blockIdx.x * 256 + threadIdx.x;
  if (t < E * C) {
    int e = t / C, c = t - e * C;
    atomicAdd(&out[(size_t)dst[e] * C + c], x[(size_t)src[e] * C + c] * ec[e]);
  }
}
__global__ void k_gather(float* __restrict__ out, const float* __restrict__ in,
                         const int* __restrict__ ids, int n, int C) {
  int t = blockIdx.x * 256 + threadIdx.x;
  if (t < n * C) {
    int r = t / C, c = t - r * C;
    out[t] = in[(size_t)ids[r] * C + c];
  }
}
__global__ void k_scatset(float* __restrict__ out, const float* __restrict__ in,
                          const int* __restrict__ ids, int n, int C) {
  int t = blockIdx.x * 256 + threadIdx.x;
  if (t < n * C) {
    int r = t / C, c = t - r * C;
    out[(size_t)ids[r] * C + c] = in[t];
  }
}
__global__ void k_add(float* __restrict__ out, const float* __restrict__ a,
                      const float* __restrict__ b, int n) {
  int t = blockIdx.x * 256 + threadIdx.x;
  if (t < n) out[t] = a[t] + b[t];
}

// ---------------------------------------------------------------------------
extern "C" void kernel_launch(void* const* d_in, const int* in_sizes, int n_in,
                              void* d_out, int out_size, void* d_ws, size_t ws_size,
                              hipStream_t stream) {
  (void)in_sizes; (void)n_in; (void)out_size; (void)ws_size;
  const int NL[5] = {50000, 25000, 12500, 6250, 3125};

  const float* x   = (const float*)d_in[0];
  const float* pos = (const float*)d_in[1];
  const int* g[5];  for (int l = 0; l < 5; ++l) g[l]   = (const int*)d_in[2 + l];
  const int* ids[4]; for (int l = 0; l < 4; ++l) ids[l] = (const int*)d_in[7 + l];

  // ---- param leaves (JAX pytree: sorted dict keys). Base indices:
  //   bottom=11 (edge 8 leaves, node 8), dec=27 (6), down[i]=33+16i,
  //   enc=97, up[i]=105+16i. MLP leaf order: W0,W1,W2,b0,b1,b2,[be,g]
  struct Mlp { const float* W[3]; const float* b[3]; const float* g; const float* be; };
  auto P = [&](int i) { return (const float*)d_in[i]; };
  auto mk = [&](int base, bool ln) {
    Mlp m;
    for (int i = 0; i < 3; ++i) { m.W[i] = P(base + i); m.b[i] = P(base + 3 + i); }
    m.be = ln ? P(base + 6) : nullptr;
    m.g  = ln ? P(base + 7) : nullptr;
    return m;
  };
  Mlp all[20]; int ind0[20];
  all[0] = mk(97, true);  ind0[0] = 10;    // enc
  all[1] = mk(27, false); ind0[1] = 64;    // dec
  all[2] = mk(11, true);  ind0[2] = 132;   // bottom edge
  all[3] = mk(19, true);  ind0[3] = 128;   // bottom node
  for (int i = 0; i < 4; ++i) {
    all[4 + 2 * i]  = mk(33 + 16 * i, true);      ind0[4 + 2 * i]  = 132; // down edge
    all[5 + 2 * i]  = mk(33 + 16 * i + 8, true);  ind0[5 + 2 * i]  = 128; // down node
    all[12 + 2 * i] = mk(105 + 16 * i, true);     ind0[12 + 2 * i] = 132; // up edge
    all[13 + 2 * i] = mk(105 + 16 * i + 8, true); ind0[13 + 2 * i] = 128; // up node
  }

  // ---- workspace layout
  char* ws = (char*)d_ws;
  size_t cur = 0;
  auto alloc = [&](size_t bytes) -> char* {
    char* p = ws + cur;
    cur += (bytes + 255) & ~(size_t)255;
    return p;
  };
  _Float16* PW = (_Float16*)alloc((size_t)60 * 5 * 2048 * sizeof(_Float16));
  float* PB = (float*)alloc((size_t)60 * 64 * sizeof(float));
  float* hA[5];   for (int l = 0; l < 5; ++l) hA[l]   = (float*)alloc((size_t)NL[l] * 64 * 4);
  float* posb[5]; for (int l = 0; l < 5; ++l) posb[l] = (float*)alloc((size_t)NL[l] * 3 * 4);
  float* aggr = (float*)alloc((size_t)NL[0] * 64 * 4);
  float* tU   = (float*)alloc((size_t)NL[0] * 64 * 4);
  float* tV   = (float*)alloc((size_t)NL[0] * 64 * 4);
  float* ecb[4]; for (int l = 0; l < 4; ++l) ecb[l] = (float*)alloc((size_t)NL[l] * 16 * 4);
  float* wlv[5]; for (int l = 0; l < 5; ++l) wlv[l] = (float*)alloc((size_t)NL[l] * 4);
  float* deg = (float*)alloc((size_t)NL[0] * 4);
  float* nrm = (float*)alloc((size_t)NL[0] * 4);
  float* aw  = (float*)alloc((size_t)NL[0] * 4);

  auto pw = [&](int m, int li) { return (const _Float16*)(PW + (size_t)(m * 3 + li) * 5 * 2048); };
  auto pb = [&](int m, int li) { return (const float*)(PB + (size_t)(m * 3 + li) * 64); };
  auto grid = [](int n) { return (n + 255) / 256; };

  // ---- pack all 60 layers (f16 fragments + padded biases)
  for (int m = 0; m < 20; ++m) {
    int c0 = (ind0[m] + 31) / 32;
    for (int li = 0; li < 3; ++li) {
      int Kreal = (li == 0) ? ind0[m] : 64;
      int Nreal = (m == 1 && li == 2) ? 3 : 64;
      int nCh = (li == 0) ? c0 : 2;
      int total = nCh * 2048;
      k_pack_w<<<grid(total), 256, 0, stream>>>(all[m].W[li], (_Float16*)pw(m, li),
                                                Kreal, Nreal, nCh);
      k_pack_b<<<1, 64, 0, stream>>>(all[m].b[li], (float*)pb(m, li), Nreal);
    }
  }

  auto dense = [&](const float* inA, int colsA, const float* inB, int colsB,
                   const float* resid, float* out, int outCols, int N,
                   int slot, int c0, int ldk, const float* gn, const float* bt) {
    int tiles = (N + 15) / 16, blocks = (tiles + 3) / 4;
    k_dense_mlp<<<blocks, 128, 0, stream>>>(inA, colsA, inB, colsB, resid, out, outCols, N,
                                            pw(slot, 0), pb(slot, 0), c0, ldk,
                                            pw(slot, 1), pb(slot, 1),
                                            pw(slot, 2), pb(slot, 2), gn, bt);
  };
  auto run_gmp = [&](int se, int sn, float* h, const float* ps, const int* gg, int E, int n) {
    hipMemsetAsync(aggr, 0, (size_t)n * 64 * sizeof(float), stream);
    int tiles = (E + 15) / 16, blocks = (tiles + 3) / 4;
    k_edge_gmp<<<blocks, 128, 0, stream>>>(h, ps, gg, gg + E, E,
                                           pw(se, 0), pb(se, 0), pw(se, 1), pb(se, 1),
                                           pw(se, 2), pb(se, 2), all[se].g, all[se].be, aggr);
    dense(h, 64, aggr, 64, h, h, 64, n, sn, 4, 128, all[sn].g, all[sn].be);
  };

  // ---- encoder, pos copy, w = ones
  dense(x, 10, nullptr, 0, nullptr, hA[0], 64, NL[0], 0, 1, 32, all[0].g, all[0].be);
  hipMemcpyAsync(posb[0], pos, (size_t)NL[0] * 3 * sizeof(float),
                 hipMemcpyDeviceToDevice, stream);
  k_fillv<<<grid(NL[0]), 256, 0, stream>>>(wlv[0], NL[0], 1.f);

  // ---- down pass
  for (int l = 0; l < 4; ++l) {
    int n = NL[l], E = n * 16;
    const int* gi = g[l];
    const int* gj = g[l] + E;
    run_gmp(4 + 2 * l, 5 + 2 * l, hA[l], posb[l], g[l], E, n);   // hA[l] == down_h[l]
    // cal_ew
    hipMemsetAsync(deg, 0, (size_t)n * 4, stream);
    k_deg<<<grid(E), 256, 0, stream>>>(gi, E, deg);
    k_normed<<<grid(n), 256, 0, stream>>>(wlv[l], deg, nrm, n);
    hipMemsetAsync(aw, 0, (size_t)n * 4, stream);
    k_aggrw<<<grid(E), 256, 0, stream>>>(gi, gj, nrm, aw, E);
    k_ec<<<grid(E), 256, 0, stream>>>(gi, gj, nrm, aw, ecb[l], E);
    k_wnew<<<grid(n), 256, 0, stream>>>(aw, deg, n);             // deg := w_full
    // weighted pool (aggregating: dst=i, src=j), then restrict by ids
    hipMemsetAsync(tU, 0, (size_t)n * 64 * 4, stream);
    k_wconv<<<grid(E * 64), 256, 0, stream>>>(hA[l], gj, gi, ecb[l], tU, E, 64);
    k_gather<<<grid(NL[l + 1] * 64), 256, 0, stream>>>(hA[l + 1], tU, ids[l], NL[l + 1], 64);
    hipMemsetAsync(tV, 0, (size_t)n * 3 * 4, stream);
    k_wconv<<<grid(E * 3), 256, 0, stream>>>(posb[l], gj, gi, ecb[l], tV, E, 3);
    k_gather<<<grid(NL[l + 1] * 3), 256, 0, stream>>>(posb[l + 1], tV, ids[l], NL[l + 1], 3);
    k_gather<<<grid(NL[l + 1]), 256, 0, stream>>>(wlv[l + 1], deg, ids[l], NL[l + 1], 1);
  }

  // ---- bottom
  run_gmp(2, 3, hA[4], posb[4], g[4], NL[4] * 16, NL[4]);

  // ---- up pass
  for (int k = 0; k < 4; ++k) {
    int l = 3 - k;
    int n = NL[l], E = n * 16;
    const int* gi = g[l];
    const int* gj = g[l] + E;
    hipMemsetAsync(tU, 0, (size_t)n * 64 * 4, stream);
    k_scatset<<<grid(NL[l + 1] * 64), 256, 0, stream>>>(tU, hA[l + 1], ids[l], NL[l + 1], 64);
    hipMemsetAsync(tV, 0, (size_t)n * 64 * 4, stream);
    k_wconv<<<grid(E * 64), 256, 0, stream>>>(tU, gi, gj, ecb[l], tV, E, 64); // dst=j (unpool)
    k_add<<<grid(n * 64), 256, 0, stream>>>(hA[l], tV, hA[l], n * 64);        // + down_h[l]
    run_gmp(12 + 2 * k, 13 + 2 * k, hA[l], posb[l], g[l], E, n);
  }

  // ---- decoder -> d_out (fp32, 50000 x 3)
  dense(hA[0], 64, nullptr, 0, nullptr, (float*)d_out, 3, NL[0], 1, 2, 64, nullptr, nullptr);
}
